// StudentKnowledgeModel_32109175505601
// MI455X (gfx1250) — compile-verified, matrix operand order checked by reading, not run on hardware
//
#include <hip/hip_runtime.h>
#include <stdint.h>

#define BB 32
#define SS 10
#define KK 4
#define DD 128
#define NQn 2001
#define NCn 501
#define NCP 512      // padded so every 16-row tile belongs to one batch b
#define NQP 2048     // padded so tiles-per-b divides into 4-wave blocks
#define LEARN_THR 2.566666666666667f
#define FORGET_CAP 7.516666666666667f

typedef _Float16 half16 __attribute__((ext_vector_type(16)));
typedef _Float16 half8  __attribute__((ext_vector_type(8)));
typedef float    f32x8  __attribute__((ext_vector_type(8)));
typedef float    f32x4  __attribute__((ext_vector_type(4)));
typedef int      i32x4  __attribute__((ext_vector_type(4)));

// CDNA5 async global->LDS copy path (guarded; falls back to sync LDS copy)
#if defined(__has_builtin)
#  if __has_builtin(__builtin_amdgcn_global_load_async_to_lds_b128) && \
      __has_builtin(__builtin_amdgcn_s_wait_asynccnt)
#    define HAVE_ASYNC_LDS 1
#  endif
#endif
#ifndef HAVE_ASYNC_LDS
#  define HAVE_ASYNC_LDS 0
#endif

#if HAVE_ASYNC_LDS
typedef __attribute__((address_space(1))) i32x4* gv4p;   // global int4*
typedef __attribute__((address_space(3))) i32x4* lv4p;   // LDS int4*
#endif

__device__ __forceinline__ float softplusf(float x) {
    return fmaxf(x, 0.f) + log1pf(expf(-fabsf(x)));
}

// ---------------------------------------------------------------------------
// Weight packer: fp32 (Kdim x Ndim, row-major, x@W convention) -> f16 B-fragments.
// Fragment layout matches ISA 7.12.2 16-bit operand layout (wave32).
// ---------------------------------------------------------------------------
__global__ void k_pack(const float* __restrict__ W, int Kdim, int Ndim,
                       unsigned short* __restrict__ out)
{
    int idx = blockIdx.x * blockDim.x + threadIdx.x;
    int total = Kdim * Ndim;
    if (idx >= total) return;
    int ntiles = Ndim >> 4;
    int s    = idx & 15;
    int frag = idx >> 4;
    int lane = frag & 31;
    int nt   = (frag >> 5) % ntiles;
    int kc   = frag / (32 * ntiles);
    int j = s >> 1, hl = s & 1;
    int k = kc * 32 + ((j < 4) ? 0 : 16) + ((lane >= 16) ? 8 : 0) + ((j & 3) << 1) + hl;
    int n = nt * 16 + (lane & 15);
    union { _Float16 h; unsigned short u; } cv;
    cv.h = (_Float16)W[(size_t)k * Ndim + n];
    out[idx] = cv.u;
}

// ---------------------------------------------------------------------------
// Big batched time-MLP: 16 state rows per wave, f16 WMMA, one path per BLOCK
// (blocks are b-major so learn/forget is block-uniform).
// Layer-1 weights (32KB) staged to LDS via async global->LDS loads.
// Layers 2/3 B-fragments double-buffered in registers over global loads.
// ---------------------------------------------------------------------------
__global__ void __launch_bounds__(128)
k_time_big(float* __restrict__ state, int rowsPerB, int blocksPerB,
           const unsigned short* __restrict__ pk1L, const unsigned short* __restrict__ pk2L,
           const unsigned short* __restrict__ pk3L,
           const unsigned short* __restrict__ pk1F, const unsigned short* __restrict__ pk2F,
           const unsigned short* __restrict__ pk3F,
           const float* __restrict__ bias1L, const float* __restrict__ bias1F,
           const float* __restrict__ b2L, const float* __restrict__ b3L,
           const float* __restrict__ b2F, const float* __restrict__ b3F,
           const float* __restrict__ ti, int sIdx)
{
    __shared__ __align__(16) unsigned short w1s[128 * 128];   // 32 KB: staged L1 weights
    __shared__ __align__(16) _Float16 h1s[4][16][128];        // 16 KB
    __shared__ __align__(16) _Float16 h2s[4][16][64];         //  8 KB

    const int lane = threadIdx.x & 31;
    const int wv   = threadIdx.x >> 5;
    const int b    = blockIdx.x / blocksPerB;
    const int tIB  = (blockIdx.x % blocksPerB) * 4 + wv;
    const int row0 = b * rowsPerB + tIB * 16;

    const float dt   = ti[b * SS + sIdx];
    const float tpos = fmaxf(dt, 0.f);
    const bool learn = tpos < LEARN_THR;     // uniform across the whole block

    const unsigned short* p1 = learn ? pk1L : pk1F;
    const half16* bp2 = (const half16*)(learn ? pk2L : pk2F);
    const half16* bp3 = (const half16*)(learn ? pk3L : pk3F);
    const float*  bb1 = learn ? bias1L : bias1F;
    const float*  bb2 = learn ? b2L : b2F;
    const float*  bb3 = learn ? b3L : b3F;

    // ---- kick off async staging of L1 weights into LDS (2048 x b128) ----
#if HAVE_ASYNC_LDS
    {
        char* gsrc = (char*)p1;
#pragma unroll
        for (int it = 0; it < 16; ++it) {
            const int e = (it * 128 + (int)threadIdx.x) * 16;   // byte offset
            __builtin_amdgcn_global_load_async_to_lds_b128(
                (gv4p)(gsrc + e), (lv4p)((char*)w1s + e), 0, 0);
        }
    }
#else
    {
#pragma unroll
        for (int it = 0; it < 16; ++it) {
            const int e = it * 128 + (int)threadIdx.x;          // b128 index
            ((f32x4*)w1s)[e] = ((const f32x4*)p1)[e];
        }
    }
#endif

    const int r   = lane & 15;
    const int hi8 = (lane >> 4) << 3;

    // ---- A fragments of the 16x128 state tile (fp32 -> fp16), overlaps copy ----
    const float* srow = state + (size_t)(row0 + r) * DD;
    half16 a[4];
#pragma unroll
    for (int kc = 0; kc < 4; ++kc) {
        const f32x4 f0 = *(const f32x4*)(srow + kc * 32 + hi8);
        const f32x4 f1 = *(const f32x4*)(srow + kc * 32 + hi8 + 4);
        const f32x4 f2 = *(const f32x4*)(srow + kc * 32 + 16 + hi8);
        const f32x4 f3 = *(const f32x4*)(srow + kc * 32 + 16 + hi8 + 4);
        half16 aa;
#pragma unroll
        for (int j = 0; j < 4; ++j) {
            aa[j]      = (_Float16)f0[j];
            aa[4 + j]  = (_Float16)f1[j];
            aa[8 + j]  = (_Float16)f2[j];
            aa[12 + j] = (_Float16)f3[j];
        }
        a[kc] = aa;
    }

#if HAVE_ASYNC_LDS
    __builtin_amdgcn_s_wait_asynccnt(0);
#endif
    __syncthreads();

    // ---- layer 1: 128 -> 128 (+per-batch t-folded bias), relu; weights in LDS ----
    const half16* bp1l = (const half16*)w1s;
#pragma unroll
    for (int nt = 0; nt < 8; ++nt) {
        f32x8 acc = {};
#pragma unroll
        for (int kc = 0; kc < 4; ++kc) {
            half16 bf = bp1l[(kc * 8 + nt) * 32 + lane];
            acc = __builtin_amdgcn_wmma_f32_16x16x32_f16(false, a[kc], false, bf,
                                                         (short)0, acc, false, false);
        }
        const float bv = bb1[b * DD + nt * 16 + r];
#pragma unroll
        for (int i = 0; i < 8; ++i)
            h1s[wv][i + hi8][nt * 16 + r] = (_Float16)fmaxf(acc[i] + bv, 0.f);
    }
    __syncthreads();

    // ---- layer 2: 128 -> 64, relu; global B-frags double-buffered ----
    half16 a2[4];
#pragma unroll
    for (int kc = 0; kc < 4; ++kc) {
        const half8 lo = *(const half8*)&h1s[wv][r][kc * 32 + hi8];
        const half8 hi = *(const half8*)&h1s[wv][r][kc * 32 + 16 + hi8];
        half16 aa;
#pragma unroll
        for (int j = 0; j < 8; ++j) { aa[j] = lo[j]; aa[8 + j] = hi[j]; }
        a2[kc] = aa;
    }
    {
        half16 c2[4], n2[4];
#pragma unroll
        for (int kc = 0; kc < 4; ++kc) c2[kc] = bp2[(kc * 4 + 0) * 32 + lane];
#pragma unroll
        for (int nt = 0; nt < 4; ++nt) {
            if (nt < 3) {
#pragma unroll
                for (int kc = 0; kc < 4; ++kc) n2[kc] = bp2[(kc * 4 + nt + 1) * 32 + lane];
            }
            f32x8 acc = {};
#pragma unroll
            for (int kc = 0; kc < 4; ++kc)
                acc = __builtin_amdgcn_wmma_f32_16x16x32_f16(false, a2[kc], false, c2[kc],
                                                             (short)0, acc, false, false);
            const float bv = bb2[nt * 16 + r];
#pragma unroll
            for (int i = 0; i < 8; ++i)
                h2s[wv][i + hi8][nt * 16 + r] = (_Float16)fmaxf(acc[i] + bv, 0.f);
#pragma unroll
            for (int kc = 0; kc < 4; ++kc) c2[kc] = n2[kc];
        }
    }
    __syncthreads();

    // ---- layer 3: 64 -> 128, double-buffered, then combine with state ----
    half16 a3[2];
#pragma unroll
    for (int kc = 0; kc < 2; ++kc) {
        const half8 lo = *(const half8*)&h2s[wv][r][kc * 32 + hi8];
        const half8 hi = *(const half8*)&h2s[wv][r][kc * 32 + 16 + hi8];
        half16 aa;
#pragma unroll
        for (int j = 0; j < 8; ++j) { aa[j] = lo[j]; aa[8 + j] = hi[j]; }
        a3[kc] = aa;
    }
    {
        half16 c3[2], n3[2];
#pragma unroll
        for (int kc = 0; kc < 2; ++kc) c3[kc] = bp3[(kc * 8 + 0) * 32 + lane];
#pragma unroll
        for (int nt = 0; nt < 8; ++nt) {
            if (nt < 7) {
#pragma unroll
                for (int kc = 0; kc < 2; ++kc) n3[kc] = bp3[(kc * 8 + nt + 1) * 32 + lane];
            }
            f32x8 acc = {};
#pragma unroll
            for (int kc = 0; kc < 2; ++kc)
                acc = __builtin_amdgcn_wmma_f32_16x16x32_f16(false, a3[kc], false, c3[kc],
                                                             (short)0, acc, false, false);
            const float bv = bb3[nt * 16 + r];
            const int d = nt * 16 + r;
#pragma unroll
            for (int i = 0; i < 8; ++i) {
                const size_t gi = (size_t)(row0 + i + hi8) * DD + d;
                const float st = state[gi];
                const float m  = acc[i] + bv;
                state[gi] = learn ? (st + m) : (st - st * (1.f / (1.f + expf(-m))));
            }
#pragma unroll
            for (int kc = 0; kc < 2; ++kc) c3[kc] = n3[kc];
        }
    }
}

// ---------------------------------------------------------------------------
// Small plain-VALU kernels for the per-event work (negligible FLOPs)
// ---------------------------------------------------------------------------
__global__ void k_zero(float* __restrict__ p, int n)
{
    int i = blockIdx.x * blockDim.x + threadIdx.x;
    if (i < n) p[i] = 0.f;
}

__global__ void k_add(float* __restrict__ a, const float* __restrict__ b, int n)
{
    int i = blockIdx.x * blockDim.x + threadIdx.x;
    if (i < n) a[i] += b[i];
}

__global__ void k_embed(const int* __restrict__ qid, const int* __restrict__ cidp,
                        const float* __restrict__ score, const int* __restrict__ err,
                        const float* __restrict__ probE, const float* __restrict__ concE,
                        const float* __restrict__ scoreE,
                        const float* __restrict__ lcorr, const float* __restrict__ syn,
                        const float* __restrict__ carelessV,
                        const float* __restrict__ lerr, const float* __restrict__ prof,
                        float* __restrict__ q_emb, float* __restrict__ c_emb,
                        float* __restrict__ s_emb,
                        float* __restrict__ logic_r, float* __restrict__ concept_r)
{
    int idx = blockIdx.x * blockDim.x + threadIdx.x;
    if (idx >= BB * SS * DD) return;
    int d = idx % DD, bs = idx / DD;
    q_emb[idx] = softplusf(probE[(size_t)qid[bs] * DD + d]);
    float sc = score[bs];
    int si = (int)(sc * 100.f);
    si = si < 0 ? 0 : (si > 100 ? 100 : si);
    s_emb[idx] = softplusf(scoreE[(size_t)si * DD + d]);
    float e0 = (float)err[bs * 5 + 0], e1 = (float)err[bs * 5 + 1];
    float e2 = (float)err[bs * 5 + 2], e3 = (float)err[bs * 5 + 3];
    float e4 = (float)err[bs * 5 + 4];
    logic_r[idx]   = e0 * lcorr[d] + e4 * lerr[d];
    concept_r[idx] = e1 * syn[d] + e2 * carelessV[d] + e3 * prof[d];
    for (int k = 0; k < KK; ++k) {
        int c = cidp[bs * KK + k];
        float v = (c == 0) ? 0.f : concE[(size_t)c * DD + d];
        c_emb[(size_t)(bs * KK + k) * DD + d] = softplusf(v);
    }
}

__global__ void k_rel_qc(const float* __restrict__ q_emb, const float* __restrict__ c_emb,
                         const int* __restrict__ cidp, float* __restrict__ rel,
                         float* __restrict__ qc)
{
    __shared__ float sraw[KK];
    __shared__ float srel[KK];
    int bs = blockIdx.x, tid = threadIdx.x;
    const float* qe = q_emb + (size_t)bs * DD;
    if (tid < KK) {
        const float* ce = c_emb + (size_t)(bs * KK + tid) * DD;
        float a = 0.f;
        for (int d2 = 0; d2 < DD; ++d2) a += ce[d2] * qe[d2];
        sraw[tid] = (cidp[bs * KK + tid] != 0) ? a : -1e9f;
    }
    __syncthreads();
    if (tid == 0) {
        float mx = sraw[0];
        for (int k = 1; k < KK; ++k) mx = fmaxf(mx, sraw[k]);
        float s = 0.f;
        for (int k = 0; k < KK; ++k) { srel[k] = expf(sraw[k] - mx); s += srel[k]; }
        for (int k = 0; k < KK; ++k) { srel[k] /= s; rel[bs * KK + k] = srel[k]; }
    }
    __syncthreads();
    float a = qe[tid];
    for (int k = 0; k < KK; ++k) a += srel[k] * c_emb[(size_t)(bs * KK + k) * DD + tid];
    qc[(size_t)bs * DD + tid] = a;
}

__global__ void k_init_state(float* __restrict__ dst, const float* __restrict__ init,
                             int rowsPerB, int validRows)
{
    long long idx = (long long)blockIdx.x * blockDim.x + threadIdx.x;
    long long total = (long long)BB * rowsPerB * DD;
    if (idx >= total) return;
    int d = (int)(idx % DD);
    int i = (int)((idx / DD) % rowsPerB);
    dst[idx] = (i < validRows) ? init[(size_t)i * DD + d] : 0.f;
}

__global__ void k_concat(float* __restrict__ dst, int rows,
                         const float* s0, int l0, int d0, long long h0, long long w0,
                         const float* s1, int l1, int d1, long long h1, long long w1,
                         const float* s2, int l2, int d2, long long h2, long long w2,
                         const float* s3, int l3, int d3, long long h3, long long w3)
{
    int L = l0 + l1 + l2 + l3;
    int idx = blockIdx.x * blockDim.x + threadIdx.x;
    if (idx >= rows * L) return;
    int r = idx / L, j = idx % L;
    const float* s; int dv; long long sh, sl;
    if (j < l0)                { s = s0; dv = d0; sh = h0; sl = w0; }
    else if (j < l0 + l1)      { j -= l0; s = s1; dv = d1; sh = h1; sl = w1; }
    else if (j < l0 + l1 + l2) { j -= l0 + l1; s = s2; dv = d2; sh = h2; sl = w2; }
    else                       { j -= l0 + l1 + l2; s = s3; dv = d3; sh = h3; sl = w3; }
    dst[idx] = s[(long long)(r / dv) * sh + (long long)(r % dv) * sl + j];
}

__global__ void k_linear(const float* __restrict__ X, const float* __restrict__ W,
                         const float* __restrict__ bias, float* __restrict__ Y,
                         int rows, int Kd, int Nd, int relu, int accum, int ostride)
{
    int idx = blockIdx.x * blockDim.x + threadIdx.x;
    if (idx >= rows * Nd) return;
    int rr = idx / Nd, o = idx % Nd;
    float acc = bias ? bias[o] : 0.f;
    const float* x = X + (size_t)rr * Kd;
    for (int k = 0; k < Kd; ++k) acc = fmaf(x[k], W[(size_t)k * Nd + o], acc);
    if (relu) acc = fmaxf(acc, 0.f);
    float* y = Y + (size_t)rr * ostride + o;
    if (accum) *y += acc; else *y = acc;
}

__global__ void k_gather_g(float* __restrict__ g, const float* __restrict__ concept,
                           const int* __restrict__ cidp, int t)
{
    int idx = blockIdx.x * blockDim.x + threadIdx.x;
    if (idx >= BB * KK * DD) return;
    int d = idx % DD, rk = idx / DD;
    int b = rk / KK, k = rk % KK;
    int c = cidp[(b * SS + t) * KK + k];
    g[idx] = concept[((size_t)b * NCP + c) * DD + d];
}

// Deterministic scatter: each thread owns (b,d), loops k serially (handles dup cids)
__global__ void k_scatter(float* __restrict__ concept, const float* __restrict__ upd,
                          const float* __restrict__ rel, const int* __restrict__ cidp, int t)
{
    int idx = blockIdx.x * blockDim.x + threadIdx.x;
    if (idx >= BB * DD) return;
    int d = idx % DD, b = idx / DD;
    for (int k = 0; k < KK; ++k) {
        int c = cidp[(b * SS + t) * KK + k];
        float w = rel[(b * SS + t) * KK + k];
        concept[((size_t)b * NCP + c) * DD + d] += upd[(size_t)(b * KK + k) * DD + d] * w;
    }
}

__global__ void k_gather_row(float* __restrict__ dst, const float* __restrict__ state,
                             int rowsPerB, const int* __restrict__ qid, int t)
{
    int idx = blockIdx.x * blockDim.x + threadIdx.x;
    if (idx >= BB * DD) return;
    int d = idx % DD, b = idx / DD;
    int q = qid[b * SS + t];
    dst[idx] = state[((size_t)b * rowsPerB + q) * DD + d];
}

__global__ void k_set_row(float* __restrict__ state, int rowsPerB,
                          const float* __restrict__ pu, const float* __restrict__ dpu,
                          const int* __restrict__ qid, int t)
{
    int idx = blockIdx.x * blockDim.x + threadIdx.x;
    if (idx >= BB * DD) return;
    int d = idx % DD, b = idx / DD;
    int q = qid[b * SS + t];
    state[((size_t)b * rowsPerB + q) * DD + d] = pu[idx] + dpu[idx];
}

__global__ void k_cstate(float* __restrict__ cs, const float* __restrict__ concept,
                         const int* __restrict__ cidp, const float* __restrict__ rel, int t)
{
    int idx = blockIdx.x * blockDim.x + threadIdx.x;
    if (idx >= BB * DD) return;
    int d = idx % DD, b = idx / DD;
    float a = 0.f;
    for (int k = 0; k < KK; ++k) {
        int c = cidp[(b * SS + t) * KK + k];
        a += rel[(b * SS + t) * KK + k] * concept[((size_t)b * NCP + c) * DD + d];
    }
    cs[idx] = a;
}

// Per-step layer-1 bias with t (or capped tc) folded from W0 row 128
__global__ void k_bias1(float* __restrict__ bias1, const float* __restrict__ ti, int sIdx,
                        const float* W0cl, const float* b0cl, const float* W0cf, const float* b0cf,
                        const float* W0pl, const float* b0pl, const float* W0pf, const float* b0pf)
{
    int idx = blockIdx.x * blockDim.x + threadIdx.x;
    if (idx >= 4 * BB * DD) return;
    int o = idx % DD;
    int b = (idx / DD) % BB;
    int m = idx / (DD * BB);
    float dt = ti[b * SS + sIdx];
    float t  = fmaxf(dt, 0.f);
    float tc = fminf(t, FORGET_CAP);
    const float* W0; const float* b0; float tv;
    if (m == 0)      { W0 = W0cl; b0 = b0cl; tv = t; }
    else if (m == 1) { W0 = W0cf; b0 = b0cf; tv = tc; }
    else if (m == 2) { W0 = W0pl; b0 = b0pl; tv = t; }
    else             { W0 = W0pf; b0 = b0pf; tv = tc; }
    bias1[idx] = b0[o] + tv * W0[128 * DD + o];
}

// time_mlp_change on the tiny logic state (32 rows) — plain VALU
__global__ void k_time_small(float* __restrict__ logic, const float* __restrict__ ti, int sIdx,
                             const float* lW0, const float* lb0, const float* lW1, const float* lb1,
                             const float* lW2, const float* lb2,
                             const float* fW0, const float* fb0, const float* fW1, const float* fb1,
                             const float* fW2, const float* fb2)
{
    __shared__ float x[129];
    __shared__ float h1[128];
    __shared__ float h2[64];
    const int b = blockIdx.x, tid = threadIdx.x;
    const float dt = ti[b * SS + sIdx];
    const float t  = fmaxf(dt, 0.f);
    const float tc = fminf(t, FORGET_CAP);
    const bool learn = t < LEARN_THR;
    const float* W0 = learn ? lW0 : fW0; const float* b0 = learn ? lb0 : fb0;
    const float* W1 = learn ? lW1 : fW1; const float* b1 = learn ? lb1 : fb1;
    const float* W2 = learn ? lW2 : fW2; const float* b2 = learn ? lb2 : fb2;
    x[tid] = logic[b * DD + tid];
    if (tid == 0) x[128] = learn ? t : tc;
    __syncthreads();
    float acc = b0[tid];
    for (int k = 0; k < 129; ++k) acc += x[k] * W0[k * 128 + tid];
    h1[tid] = fmaxf(acc, 0.f);
    __syncthreads();
    if (tid < 64) {
        float a2 = b1[tid];
        for (int k = 0; k < 128; ++k) a2 += h1[k] * W1[k * 64 + tid];
        h2[tid] = fmaxf(a2, 0.f);
    }
    __syncthreads();
    float a3 = b2[tid];
    for (int k = 0; k < 64; ++k) a3 += h2[k] * W2[k * 128 + tid];
    const float st = x[tid];
    logic[b * DD + tid] = learn ? (st + a3) : (st - st * (1.f / (1.f + expf(-a3))));
}

__global__ void k_predict(const float* __restrict__ la, const float* __restrict__ ca,
                          const float* __restrict__ pa,
                          const float* __restrict__ ability_w, const float* __restrict__ disc,
                          const float* __restrict__ diff,
                          const int* __restrict__ qid, int sOut, float* __restrict__ outScore)
{
    int b = threadIdx.x;
    if (b >= BB) return;
    int q = qid[b * SS + sOut];
    float w0 = ability_w[q * 3 + 0], w1 = ability_w[q * 3 + 1], w2 = ability_w[q * 3 + 2];
    float mx = fmaxf(w0, fmaxf(w1, w2));
    float e0 = expf(w0 - mx), e1 = expf(w1 - mx), e2 = expf(w2 - mx);
    float s = e0 + e1 + e2;
    float ability = (e0 * la[b] + e1 * ca[b] + e2 * pa[b]) / s;
    float z = disc[q] * (ability - diff[q]);
    outScore[b * SS + sOut] = 1.f / (1.f + expf(-z));
}

// ---------------------------------------------------------------------------
// Host-side helpers
// ---------------------------------------------------------------------------
struct Src { const float* p; int len; int div; long long sh; long long sl; };

static inline int ceilDiv(int a, int b) { return (a + b - 1) / b; }

static void concat4(hipStream_t st, float* dst, int rows, Src a, Src b, Src c, Src d)
{
    int n = rows * (a.len + b.len + c.len + d.len);
    k_concat<<<ceilDiv(n, 256), 256, 0, st>>>(dst, rows,
        a.p, a.len, a.div, a.sh, a.sl,  b.p, b.len, b.div, b.sh, b.sl,
        c.p, c.len, c.div, c.sh, c.sl,  d.p, d.len, d.div, d.sh, d.sl);
}

static void lin(hipStream_t st, const float* X, const float* W, const float* b, float* Y,
                int rows, int Kd, int Nd, int relu, int accum, int ostride)
{
    int n = rows * Nd;
    k_linear<<<ceilDiv(n, 256), 256, 0, st>>>(X, W, b, Y, rows, Kd, Nd, relu, accum, ostride);
}

extern "C" void kernel_launch(void* const* d_in, const int* in_sizes, int n_in,
                              void* d_out, int out_size, void* d_ws, size_t ws_size,
                              hipStream_t stream)
{
    (void)in_sizes; (void)n_in; (void)ws_size;
    const int*   qid   = (const int*)d_in[0];
    const int*   cid   = (const int*)d_in[1];
    const float* score = (const float*)d_in[2];
    const int*   err   = (const int*)d_in[3];
    const float* ti    = (const float*)d_in[4];
    auto F = [&](int i) { return (const float*)d_in[i]; };

    // --------------- workspace carve ---------------
    char* base = (char*)d_ws;
    size_t off = 0;
    auto carve = [&](size_t bytes) -> char* {
        char* p = base + off;
        off = (off + bytes + 255) & ~(size_t)255;
        return p;
    };
    float* logic     = (float*)carve((size_t)BB * DD * 4);
    float* concept   = (float*)carve((size_t)BB * NCP * DD * 4);
    float* problem   = (float*)carve((size_t)BB * NQP * DD * 4);
    float* q_emb     = (float*)carve((size_t)BB * SS * DD * 4);
    float* s_emb     = (float*)carve((size_t)BB * SS * DD * 4);
    float* c_emb     = (float*)carve((size_t)BB * SS * KK * DD * 4);
    float* logic_r   = (float*)carve((size_t)BB * SS * DD * 4);
    float* concept_r = (float*)carve((size_t)BB * SS * DD * 4);
    float* qc        = (float*)carve((size_t)BB * SS * DD * 4);
    float* rel       = (float*)carve((size_t)BB * SS * KK * 4);
    float* bias1     = (float*)carve((size_t)4 * BB * DD * 4);
    float* g         = (float*)carve((size_t)BB * KK * DD * 4);
    float* upd       = (float*)carve((size_t)BB * KK * DD * 4);
    float* Xc        = (float*)carve((size_t)128 * 512 * 4);
    float* tmp1      = (float*)carve((size_t)128 * DD * 4);
    float* delta     = (float*)carve((size_t)BB * DD * 4);
    float* dpu       = (float*)carve((size_t)BB * DD * 4);
    float* pu        = (float*)carve((size_t)BB * DD * 4);
    float* pun       = (float*)carve((size_t)BB * DD * 4);
    float* cstate    = (float*)carve((size_t)BB * DD * 4);
    float* abilH     = (float*)carve((size_t)BB * DD * 4);
    float* laB       = (float*)carve((size_t)BB * 4);
    float* caB       = (float*)carve((size_t)BB * 4);
    float* paB       = (float*)carve((size_t)BB * 4);
    unsigned short* pkCL1 = (unsigned short*)carve(128 * 128 * 2);
    unsigned short* pkCL2 = (unsigned short*)carve(128 * 64 * 2);
    unsigned short* pkCL3 = (unsigned short*)carve(64 * 128 * 2);
    unsigned short* pkCF1 = (unsigned short*)carve(128 * 128 * 2);
    unsigned short* pkCF2 = (unsigned short*)carve(128 * 64 * 2);
    unsigned short* pkCF3 = (unsigned short*)carve(64 * 128 * 2);
    unsigned short* pkPL1 = (unsigned short*)carve(128 * 128 * 2);
    unsigned short* pkPL2 = (unsigned short*)carve(128 * 64 * 2);
    unsigned short* pkPL3 = (unsigned short*)carve(64 * 128 * 2);
    unsigned short* pkPF1 = (unsigned short*)carve(128 * 128 * 2);
    unsigned short* pkPF2 = (unsigned short*)carve(128 * 64 * 2);
    unsigned short* pkPF3 = (unsigned short*)carve(64 * 128 * 2);

    // --------------- setup ---------------
    k_zero<<<ceilDiv(out_size, 256), 256, 0, stream>>>((float*)d_out, out_size);
    k_embed<<<ceilDiv(BB * SS * DD, 256), 256, 0, stream>>>(
        qid, cid, score, err, F(5), F(6), F(7),
        F(8), F(9), F(10), F(11), F(12),
        q_emb, c_emb, s_emb, logic_r, concept_r);
    k_rel_qc<<<BB * SS, 128, 0, stream>>>(q_emb, c_emb, cid, rel, qc);
    k_init_state<<<ceilDiv(BB * DD, 256), 256, 0, stream>>>(logic, F(17), 1, 1);
    k_init_state<<<ceilDiv(BB * NCP * DD, 256), 256, 0, stream>>>(concept, F(16), NCP, NCn);
    k_init_state<<<ceilDiv(BB * NQP * DD, 256), 256, 0, stream>>>(problem, F(18), NQP, NQn);

    auto pack = [&](const float* W, int Kd, int Nd, unsigned short* o) {
        int n = Kd * Nd;
        k_pack<<<ceilDiv(n, 256), 256, 0, stream>>>(W, Kd, Nd, o);
    };
    pack(F(45), 128, 128, pkCL1); pack(F(47), 128, 64, pkCL2); pack(F(49), 64, 128, pkCL3);
    pack(F(51), 128, 128, pkCF1); pack(F(53), 128, 64, pkCF2); pack(F(55), 64, 128, pkCF3);
    pack(F(57), 128, 128, pkPL1); pack(F(59), 128, 64, pkPL2); pack(F(61), 64, 128, pkPL3);
    pack(F(63), 128, 128, pkPF1); pack(F(65), 128, 64, pkPF2); pack(F(67), 64, 128, pkPF3);

    const Src NONE { nullptr, 0, 1, 0, 0 };
    const int cBlocksPerB = (NCP / 16) / 4;   // 8
    const int pBlocksPerB = (NQP / 16) / 4;   // 32

    // --------------- sequential scan ---------------
    for (int t = 0; t < SS - 1; ++t) {
        const int sOut = t + 1;
        Src L   { logic,               DD, 1,  DD,                      0 };
        Src QC  { qc + t * DD,         DD, 1,  (long long)SS * DD,      0 };
        Src SE  { s_emb + t * DD,      DD, 1,  (long long)SS * DD,      0 };
        Src RL  { logic_r + t * DD,    DD, 1,  (long long)SS * DD,      0 };
        Src G   { g,                   DD, 1,  DD,                      0 };
        Src CE  { c_emb + t * KK * DD, DD, KK, (long long)SS * KK * DD, DD };
        Src SEk { s_emb + t * DD,      DD, KK, (long long)SS * DD,      0 };
        Src RCk { concept_r + t * DD,  DD, KK, (long long)SS * DD,      0 };

        // logic += mlp(logic_change,[logic,qc,s]) + mlp(logic_change2,[logic,qc,rl])
        concat4(stream, Xc, BB, L, QC, SE, NONE);
        lin(stream, Xc, F(69), F(70), tmp1, BB, 3 * DD, DD, 1, 0, DD);
        lin(stream, tmp1, F(71), F(72), delta, BB, DD, DD, 0, 0, DD);
        concat4(stream, Xc, BB, L, QC, RL, NONE);
        lin(stream, Xc, F(73), F(74), tmp1, BB, 3 * DD, DD, 1, 0, DD);
        lin(stream, tmp1, F(75), F(76), delta, BB, DD, DD, 0, 1, DD);
        k_add<<<ceilDiv(BB * DD, 256), 256, 0, stream>>>(logic, delta, BB * DD);

        // concept[cid] += rel * (mlp(cc,[g,ce,s]) + mlp(cc2,[g,ce,rc]))
        k_gather_g<<<ceilDiv(BB * KK * DD, 256), 256, 0, stream>>>(g, concept, cid, t);
        concat4(stream, Xc, BB * KK, G, CE, SEk, NONE);
        lin(stream, Xc, F(77), F(78), tmp1, BB * KK, 3 * DD, DD, 1, 0, DD);
        lin(stream, tmp1, F(79), F(80), upd, BB * KK, DD, DD, 0, 0, DD);
        concat4(stream, Xc, BB * KK, G, CE, RCk, NONE);
        lin(stream, Xc, F(81), F(82), tmp1, BB * KK, 3 * DD, DD, 1, 0, DD);
        lin(stream, tmp1, F(83), F(84), upd, BB * KK, DD, DD, 0, 1, DD);
        k_scatter<<<ceilDiv(BB * DD, 256), 256, 0, stream>>>(concept, upd, rel, cid, t);

        // problem[qid] = pu + mlp(pu,[pu,qc]) + mlp(pu2,[pu,qc,s])
        k_gather_row<<<ceilDiv(BB * DD, 256), 256, 0, stream>>>(pu, problem, NQP, qid, t);
        Src PU { pu, DD, 1, DD, 0 };
        concat4(stream, Xc, BB, PU, QC, NONE, NONE);
        lin(stream, Xc, F(85), F(86), tmp1, BB, 2 * DD, DD, 1, 0, DD);
        lin(stream, tmp1, F(87), F(88), dpu, BB, DD, DD, 0, 0, DD);
        concat4(stream, Xc, BB, PU, QC, SE, NONE);
        lin(stream, Xc, F(89), F(90), tmp1, BB, 3 * DD, DD, 1, 0, DD);
        lin(stream, tmp1, F(91), F(92), dpu, BB, DD, DD, 0, 1, DD);
        k_set_row<<<ceilDiv(BB * DD, 256), 256, 0, stream>>>(problem, NQP, pu, dpu, qid, t);

        // time decay/growth
        k_time_small<<<BB, 128, 0, stream>>>(logic, ti, sOut,
            F(33), F(34), F(35), F(36), F(37), F(38),
            F(39), F(40), F(41), F(42), F(43), F(44));
        k_bias1<<<ceilDiv(4 * BB * DD, 256), 256, 0, stream>>>(bias1, ti, sOut,
            F(45), F(46), F(51), F(52), F(57), F(58), F(63), F(64));
        k_time_big<<<BB * cBlocksPerB, 128, 0, stream>>>(concept, NCP, cBlocksPerB,
            pkCL1, pkCL2, pkCL3, pkCF1, pkCF2, pkCF3,
            bias1 + 0 * BB * DD, bias1 + 1 * BB * DD,
            F(48), F(50), F(54), F(56), ti, sOut);
        k_time_big<<<BB * pBlocksPerB, 128, 0, stream>>>(problem, NQP, pBlocksPerB,
            pkPL1, pkPL2, pkPL3, pkPF1, pkPF2, pkPF3,
            bias1 + 2 * BB * DD, bias1 + 3 * BB * DD,
            F(60), F(62), F(66), F(68), ti, sOut);

        // predictions at s = t+1
        k_cstate<<<ceilDiv(BB * DD, 256), 256, 0, stream>>>(cstate, concept, cid, rel, sOut);
        k_gather_row<<<ceilDiv(BB * DD, 256), 256, 0, stream>>>(pun, problem, NQP, qid, sOut);
        lin(stream, logic, F(19), F(20), abilH, BB, DD, 64, 1, 0, 64);
        lin(stream, abilH, F(21), F(22), laB, BB, 64, 1, 0, 0, 1);
        lin(stream, cstate, F(23), F(24), abilH, BB, DD, 64, 1, 0, 64);
        lin(stream, abilH, F(25), F(26), caB, BB, 64, 1, 0, 0, 1);
        Src PUN { pun, DD, 1, DD, 0 };
        Src QCN { qc + sOut * DD, DD, 1, (long long)SS * DD, 0 };
        concat4(stream, Xc, BB, PUN, QCN, NONE, NONE);
        lin(stream, Xc, F(27), F(28), tmp1, BB, 2 * DD, DD, 1, 0, DD);
        lin(stream, tmp1, F(29), F(30), abilH, BB, DD, 64, 1, 0, 64);
        lin(stream, abilH, F(31), F(32), paB, BB, 64, 1, 0, 0, 1);
        k_predict<<<1, 32, 0, stream>>>(laB, caB, paB, F(15), F(14), F(13),
                                        qid, sOut, (float*)d_out);
        Src CS { cstate, DD, 1, DD, 0 };
        concat4(stream, Xc, BB, L, CS, PUN, QCN);
        lin(stream, Xc, F(93), F(94), tmp1, BB, 4 * DD, DD, 1, 0, DD);
        lin(stream, tmp1, F(95), F(96), (float*)d_out + BB * SS + sOut * 2,
            BB, DD, 2, 0, 0, SS * 2);
        lin(stream, Xc, F(97), F(98), tmp1, BB, 4 * DD, DD, 1, 0, DD);
        lin(stream, tmp1, F(99), F(100), (float*)d_out + BB * SS + BB * SS * 2 + sOut * 3,
            BB, DD, 3, 0, 0, SS * 3);
    }
}
